// CausalSelfAttention_43353399886345
// MI455X (gfx1250) — matrix operand order repaired and checked by reading, not compiled, not run on hardware
//
#include <hip/hip_runtime.h>
#include <stdint.h>

typedef __attribute__((ext_vector_type(16))) __bf16 bf16x16;
typedef __attribute__((ext_vector_type(8)))  float  f32x8;

#define T_SEQ 4096
#define C_DIM 768
#define HD    64
#define N_QKV 2304

__device__ __forceinline__ unsigned short f2bf(float f) {
  union { float f; unsigned int u; } x; x.f = f;
  unsigned int u = x.u;
  unsigned int r = u + 0x7FFFu + ((u >> 16) & 1u);   // round-to-nearest-even
  return (unsigned short)(r >> 16);
}

__device__ __forceinline__ f32x8 wmma_bf16(const unsigned int* a, const unsigned int* b, f32x8 c) {
  union { unsigned int u[8]; bf16x16 v; } ua, ub;
#pragma unroll
  for (int i = 0; i < 8; ++i) { ua.u[i] = a[i]; ub.u[i] = b[i]; }
  return __builtin_amdgcn_wmma_f32_16x16x32_bf16(false, ua.v, false, ub.v,
                                                 (short)0, c, false, false);
}

// LDS byte offset (workgroup-relative) of a generic pointer into __shared__.
typedef const __attribute__((address_space(3))) unsigned short* lds_cu16;
__device__ __forceinline__ unsigned ldsOff(const unsigned short* p) {
  return (unsigned)(unsigned long long)(lds_cu16)p;
}

// CDNA5 async global->LDS copy, 16B per lane, tracked by ASYNCcnt.
__device__ __forceinline__ void async_load_b128(unsigned lds_byte_off, const void* gaddr) {
  asm volatile("global_load_async_to_lds_b128 %0, %1, off"
               :: "v"(lds_byte_off), "v"(gaddr) : "memory");
}
__device__ __forceinline__ void wait_async0() {
  asm volatile("s_wait_asynccnt 0x0" ::: "memory");
}

// ---------------------------------------------------------------------------
// K0: fp32 -> bf16 conversions; weights transposed to [N][K] so WMMA
// B-fragments read two consecutive K values per 32-bit register.
// ---------------------------------------------------------------------------
__global__ void prep_kernel(const float* __restrict__ x,
                            const float* __restrict__ wqkv,
                            const float* __restrict__ wproj,
                            unsigned short* __restrict__ xb,
                            unsigned short* __restrict__ wqT,
                            unsigned short* __restrict__ wpT) {
  const int NX  = T_SEQ * C_DIM;
  const int NWQ = C_DIM * N_QKV;
  const int NWP = C_DIM * C_DIM;
  int i = blockIdx.x * blockDim.x + threadIdx.x;
  if (i < NX) { xb[i] = f2bf(x[i]); return; }
  i -= NX;
  if (i < NWQ) {
    int k = i / N_QKV, n = i - k * N_QKV;
    wqT[n * C_DIM + k] = f2bf(wqkv[i]);
    return;
  }
  i -= NWQ;
  if (i < NWP) {
    int k = i / C_DIM, n = i - k * C_DIM;
    wpT[n * C_DIM + k] = f2bf(wproj[i]);
  }
}

// ---------------------------------------------------------------------------
// GEMM core: 64(M) x 128(N) block tile, K-step 64, double-buffered LDS fed by
// async global->LDS copies; 8 waves (4M x 2N); pad stride 72 (16B aligned,
// conflict-free fragment gathers). Epilogue is a lambda:
//   epi(acc, m0, n0, mi, ni, hh, ln)
// ---------------------------------------------------------------------------
template <typename Epilogue>
__device__ __forceinline__ void gemm_core(const unsigned short* __restrict__ Aptr,
                                          const unsigned short* __restrict__ Bptr,
                                          Epilogue epi) {
  __shared__ unsigned short As[2][64 * 72];
  __shared__ unsigned short Bs[2][128 * 72];
  const int tid  = threadIdx.x;
  const int wave = tid >> 5, lane = tid & 31;
  const int mi = wave & 3, ni = wave >> 2;
  const int hh = lane >> 4, ln = lane & 15;
  const int m0 = blockIdx.y * 64;
  const int n0 = blockIdx.x * 128;
  const unsigned asOff = ldsOff(&As[0][0]);
  const unsigned bsOff = ldsOff(&Bs[0][0]);
  f32x8 acc[4];
#pragma unroll
  for (int nt = 0; nt < 4; ++nt)
#pragma unroll
    for (int e = 0; e < 8; ++e) acc[nt][e] = 0.0f;

  const int srow = tid >> 3, scol = (tid & 7) * 8;
  auto stage = [&](int k0, int buf) {
#pragma unroll
    for (int i = 0; i < 2; ++i) {            // A: 64x64
      const int r = srow + i * 32;
      async_load_b128(asOff + (unsigned)((buf * 64 * 72 + r * 72 + scol) * 2),
                      &Aptr[(m0 + r) * C_DIM + k0 + scol]);
    }
#pragma unroll
    for (int i = 0; i < 4; ++i) {            // B: 128x64
      const int r = srow + i * 32;
      async_load_b128(bsOff + (unsigned)((buf * 128 * 72 + r * 72 + scol) * 2),
                      &Bptr[(n0 + r) * C_DIM + k0 + scol]);
    }
  };

  stage(0, 0);
  wait_async0();
  __syncthreads();

  for (int kt = 0; kt < C_DIM / 64; ++kt) {
    const int cur = kt & 1;
    if (kt + 1 < C_DIM / 64) stage((kt + 1) * 64, 1 - cur);
#pragma unroll
    for (int s = 0; s < 2; ++s) {
      unsigned int af[8];
      const unsigned short* Ab = &As[cur][(mi * 16 + ln) * 72 + s * 32];
#pragma unroll
      for (int v = 0; v < 8; ++v)
        af[v] = *(const unsigned int*)&Ab[(v << 1) + (hh << 4)];
      unsigned int bfr[4][8];
#pragma unroll
      for (int nt = 0; nt < 4; ++nt) {
        const unsigned short* Bb =
            &Bs[cur][(ni * 64 + nt * 16 + ln) * 72 + s * 32 + (hh << 4)];
#pragma unroll
        for (int v = 0; v < 8; ++v)
          bfr[nt][v] = *(const unsigned int*)&Bb[v << 1];
      }
#pragma unroll
      for (int nt = 0; nt < 4; ++nt)
        acc[nt] = wmma_bf16(af, bfr[nt], acc[nt]);
    }
    wait_async0();
    __syncthreads();
  }
  epi(acc, m0, n0, mi, ni, hh, ln);
}

// ---------------------------------------------------------------------------
// K1: qkv = x @ w_qkv + b_qkv -> head-major bf16 Q[h][t][d], K[h][t][d],
//     Vt[h][d][t] (Vt transposed so P*V B-fragments read K-contiguous pairs).
// ---------------------------------------------------------------------------
__global__ __launch_bounds__(256) void gemm_qkv_kernel(
    const unsigned short* __restrict__ xb,
    const unsigned short* __restrict__ wqT,
    const float* __restrict__ bqkv,
    unsigned short* __restrict__ q_rm,
    unsigned short* __restrict__ k_rm,
    unsigned short* __restrict__ v_t) {
  gemm_core(xb, wqT,
    [&](f32x8 (&acc)[4], int m0, int n0, int mi, int ni, int hh, int ln) {
#pragma unroll
      for (int nt = 0; nt < 4; ++nt) {
#pragma unroll
        for (int r = 0; r < 8; ++r) {
          const int row = m0 + mi * 16 + r + hh * 8;
          const int col = n0 + ni * 64 + nt * 16 + ln;
          const float val = acc[nt][r] + bqkv[col];
          const int which = col / C_DIM;
          const int c = col - which * C_DIM;
          const int h = c >> 6;
          const int d = c & 63;
          const unsigned short bv = f2bf(val);
          if (which == 0)      q_rm[(h * T_SEQ + row) * HD + d] = bv;
          else if (which == 1) k_rm[(h * T_SEQ + row) * HD + d] = bv;
          else                 v_t[(h * HD + d) * T_SEQ + row] = bv;
        }
      }
    });
}

// ---------------------------------------------------------------------------
// K3: out = attn @ w_proj + b_proj (fp32 output).
// ---------------------------------------------------------------------------
__global__ __launch_bounds__(256) void gemm_proj_kernel(
    const unsigned short* __restrict__ attn,
    const unsigned short* __restrict__ wpT,
    const float* __restrict__ bproj,
    float* __restrict__ out) {
  gemm_core(attn, wpT,
    [&](f32x8 (&acc)[4], int m0, int n0, int mi, int ni, int hh, int ln) {
#pragma unroll
      for (int nt = 0; nt < 4; ++nt) {
#pragma unroll
        for (int r = 0; r < 8; ++r) {
          const int row = m0 + mi * 16 + r + hh * 8;
          const int col = n0 + ni * 64 + nt * 16 + ln;
          out[row * C_DIM + col] = acc[nt][r] + bproj[col];
        }
      }
    });
}

// ---------------------------------------------------------------------------
// K2: flash attention. Block = 1 head x 128 query rows (8 waves x 16 rows).
// K/V tiles for each 32-key block are staged ONCE per block into
// double-buffered LDS via async copies (prefetching the next block while the
// current one is consumed). Uniform trip count; waves past their causal limit
// skip compute but keep barriers.
// ---------------------------------------------------------------------------
__global__ __launch_bounds__(256) void attn_kernel(
    const unsigned short* __restrict__ q_rm,
    const unsigned short* __restrict__ k_rm,
    const unsigned short* __restrict__ v_t,
    unsigned short* __restrict__ attn_out) {
  __shared__ unsigned short Ks[2][32 * 72];   // [key][d], pad 72
  __shared__ unsigned short Vs[2][64 * 40];   // [d][key], pad 40
  __shared__ unsigned short Plds[8 * 16 * 40];
  const int tid  = threadIdx.x;
  const int wave = tid >> 5, lane = tid & 31;
  const int hh = lane >> 4, ln = lane & 15;
  const int h  = blockIdx.x;
  const int q0 = blockIdx.y * 128 + wave * 16;
  const float fscale = 0.18033688011112042f;     // (1/sqrt(64)) * log2(e)
  const unsigned ksOff = ldsOff(&Ks[0][0]);
  const unsigned vsOff = ldsOff(&Vs[0][0]);

  auto stageKV = [&](int kb, int buf) {
    { const int r = tid >> 3, c = (tid & 7) * 8;                 // 32x64
      async_load_b128(ksOff + (unsigned)((buf * 32 * 72 + r * 72 + c) * 2),
                      &k_rm[(h * T_SEQ + kb + r) * HD + c]); }
    { const int r = tid >> 2, c = (tid & 3) * 8;                 // 64x32
      async_load_b128(vsOff + (unsigned)((buf * 64 * 40 + r * 40 + c) * 2),
                      &v_t[(h * HD + r) * T_SEQ + kb + c]); }
  };

  stageKV(0, 0);

  unsigned int Qf[2][8];
  const unsigned short* qbase = q_rm + (h * T_SEQ + q0 + ln) * HD;
#pragma unroll
  for (int s = 0; s < 2; ++s)
#pragma unroll
    for (int v = 0; v < 8; ++v) {
      const int k = (v << 1) + (hh << 4);
      Qf[s][v] = *(const unsigned int*)&qbase[s * 32 + k];
    }

  float m_run[8], l_run[8];
  f32x8 O[4];
#pragma unroll
  for (int r = 0; r < 8; ++r) { m_run[r] = -__builtin_inff(); l_run[r] = 0.0f; }
#pragma unroll
  for (int nt = 0; nt < 4; ++nt)
#pragma unroll
    for (int e = 0; e < 8; ++e) O[nt][e] = 0.0f;

  unsigned short* myP = &Plds[wave * 16 * 40];
  const int my_nkb  = (q0 + 47) >> 5;                     // this wave's blocks
  const int blk_nkb = (blockIdx.y * 128 + 112 + 47) >> 5; // block-uniform max

  wait_async0();
  __syncthreads();

  for (int kbi = 0; kbi < blk_nkb; ++kbi) {
    const int cur = kbi & 1;
    const int kb  = kbi * 32;
    if (kbi + 1 < blk_nkb) stageKV((kbi + 1) * 32, 1 - cur);

    if (kbi < my_nkb) {
      f32x8 S[2];
#pragma unroll
      for (int nt = 0; nt < 2; ++nt)
#pragma unroll
        for (int e = 0; e < 8; ++e) S[nt][e] = 0.0f;

#pragma unroll
      for (int nt = 0; nt < 2; ++nt) {
        unsigned int Bf[2][8];
#pragma unroll
        for (int s = 0; s < 2; ++s)
#pragma unroll
          for (int v = 0; v < 8; ++v)
            Bf[s][v] = *(const unsigned int*)
                &Ks[cur][(nt * 16 + ln) * 72 + s * 32 + (hh << 4) + (v << 1)];
        S[nt] = wmma_bf16(Qf[0], Bf[0], S[nt]);
        S[nt] = wmma_bf16(Qf[1], Bf[1], S[nt]);
      }
      // scale (log2 domain); mask only on the diagonal block
      if (kb + 31 > q0) {
#pragma unroll
        for (int nt = 0; nt < 2; ++nt)
#pragma unroll
          for (int r = 0; r < 8; ++r) {
            const int col = kb + nt * 16 + ln;
            const int row = q0 + r + hh * 8;
            const float sv = S[nt][r] * fscale;
            S[nt][r] = (col <= row) ? sv : -__builtin_inff();
          }
      } else {
#pragma unroll
        for (int nt = 0; nt < 2; ++nt)
#pragma unroll
          for (int r = 0; r < 8; ++r) S[nt][r] *= fscale;
      }
      // online softmax (reductions inside 16-lane halves match C-layout)
#pragma unroll
      for (int r = 0; r < 8; ++r) {
        float mx = fmaxf(S[0][r], S[1][r]);
#pragma unroll
        for (int msk = 8; msk >= 1; msk >>= 1)
          mx = fmaxf(mx, __shfl_xor(mx, msk, 32));
        const float nm = fmaxf(m_run[r], mx);
        const float alpha = exp2f(m_run[r] - nm);
        m_run[r] = nm;
        const float p0 = exp2f(S[0][r] - nm);
        const float p1 = exp2f(S[1][r] - nm);
        S[0][r] = p0; S[1][r] = p1;
        float rs = p0 + p1;
#pragma unroll
        for (int msk = 8; msk >= 1; msk >>= 1)
          rs += __shfl_xor(rs, msk, 32);
        l_run[r] = l_run[r] * alpha + rs;
#pragma unroll
        for (int nt = 0; nt < 4; ++nt) O[nt][r] *= alpha;
      }
      // P: C-layout -> per-wave LDS tile -> A-layout fragment
#pragma unroll
      for (int nt = 0; nt < 2; ++nt)
#pragma unroll
        for (int r = 0; r < 8; ++r)
          myP[(r + hh * 8) * 40 + nt * 16 + ln] = f2bf(S[nt][r]);
      asm volatile("s_wait_dscnt 0x0" ::: "memory");
      unsigned int Pf[8];
#pragma unroll
      for (int v = 0; v < 8; ++v) {
        const int k = (v << 1) + (hh << 4);
        Pf[v] = *(const unsigned int*)&myP[ln * 40 + k];
      }
      // O += P * V
      unsigned int Vf[4][8];
#pragma unroll
      for (int nt = 0; nt < 4; ++nt)
#pragma unroll
        for (int v = 0; v < 8; ++v)
          Vf[nt][v] = *(const unsigned int*)
              &Vs[cur][(nt * 16 + ln) * 40 + (hh << 4) + (v << 1)];
#pragma unroll
      for (int nt = 0; nt < 4; ++nt)
        O[nt] = wmma_bf16(Pf, Vf[nt], O[nt]);
    }
    wait_async0();
    __syncthreads();
  }
  // finalize
#pragma unroll
  for (int r = 0; r < 8; ++r) {
    const float inv = 1.0f / l_run[r];
    const int row = q0 + r + hh * 8;
#pragma unroll
    for (int nt = 0; nt < 4; ++nt) {
      const int col = h * HD + nt * 16 + ln;
      attn_out[row * C_DIM + col] = f2bf(O[nt][r] * inv);
    }
  }
}

// ---------------------------------------------------------------------------
extern "C" void kernel_launch(void* const* d_in, const int* in_sizes, int n_in,
                              void* d_out, int out_size, void* d_ws, size_t ws_size,
                              hipStream_t stream) {
  const float* x      = (const float*)d_in[0];
  const float* w_qkv  = (const float*)d_in[1];
  const float* b_qkv  = (const float*)d_in[2];
  const float* w_proj = (const float*)d_in[3];
  const float* b_proj = (const float*)d_in[4];

  char* ws = (char*)d_ws;
  unsigned short* xb   = (unsigned short*)(ws + 0);          //  6,291,456 B
  unsigned short* wqT  = (unsigned short*)(ws + 6291456);    //  3,538,944 B
  unsigned short* wpT  = (unsigned short*)(ws + 9830400);    //  1,179,648 B
  unsigned short* q_rm = (unsigned short*)(ws + 11010048);   //  6,291,456 B
  unsigned short* k_rm = (unsigned short*)(ws + 17301504);   //  6,291,456 B
  unsigned short* v_t  = (unsigned short*)(ws + 23592960);   //  6,291,456 B
  unsigned short* att  = (unsigned short*)(ws + 29884416);   //  6,291,456 B

  const int total = T_SEQ * C_DIM + C_DIM * N_QKV + C_DIM * C_DIM;
  prep_kernel<<<(total + 255) / 256, 256, 0, stream>>>(x, w_qkv, w_proj, xb, wqT, wpT);
  gemm_qkv_kernel<<<dim3(N_QKV / 128, T_SEQ / 64), 256, 0, stream>>>(
      xb, wqT, b_qkv, q_rm, k_rm, v_t);
  attn_kernel<<<dim3(12, T_SEQ / 128), 256, 0, stream>>>(q_rm, k_rm, v_t, att);
  gemm_proj_kernel<<<dim3(C_DIM / 128, T_SEQ / 64), 256, 0, stream>>>(
      att, wpT, b_proj, (float*)d_out);
}
